// VisionTransformer_55662776156313
// MI455X (gfx1250) — matrix operand order repaired and checked by reading, not compiled, not run on hardware
//
#include <hip/hip_runtime.h>
#include <hip/hip_bf16.h>
#include <math.h>

// ---------------------------------------------------------------------------
// ViT-Base/16 forward, B=64, on gfx1250 (MI455X).
// bf16 WMMA (v_wmma_f32_16x16x32_bf16) GEMMs, fp32 residual stream.
// Double-buffered LDS staging via global_load_async_to_lds_b128 (ASYNCcnt).
// ---------------------------------------------------------------------------

typedef __bf16 bf16;
typedef __attribute__((ext_vector_type(16))) __bf16 v16bf;
typedef __attribute__((ext_vector_type(8)))  float  v8f;

union FragU { uint4 q[2]; v16bf v; };

#define BM 128
#define BN 128
#define BK 64
#define KPAD 8   // LDS row pad (pitch 72 elems = 144B) to spread banks

// Model constants
static constexpr int  BSZ  = 64;
static constexpr int  NP   = 256;   // tokens padded 197 -> 256 (multiple of BK)
static constexpr int  NT   = 197;   // real tokens
static constexpr int  DIM  = 768;
static constexpr int  NH   = 12;
static constexpr int  HDIM = 64;
static constexpr int  MLPD = 3072;
static constexpr int  NCLS = 1000;
static constexpr int  NLAYER = 12;
static constexpr long MT   = (long)BSZ * NP;  // 16384 token rows

// ---------------------------------------------------------------------------
// Generic batched GEMM:  out = epilogue( A[M,K] @ op(B)[K,N] + bias )
//   A: bf16 row-major, row stride lda, batch offset (z/aC)*aO + (z%aC)*aI
//   B: bf16. TB==0 : W[N,K] row-major (compute A @ W^T), async-staged
//            TB==1 : W[K,N] row-major (compute A @ W), scalar transpose stage
//   modes: 0 = fp32 store (+bias)
//          1 = bf16 store (+bias)
//          2 = bf16 store of exact-erf GELU(acc+bias)
//          3 = fp32 store of outF + acc + bias   (residual accumulate)
//          4 = fp32 store of acc*scale, columns n>=maskN forced to -1e30
// Requirements: K % 64 == 0, lda/ldb % 8 == 0, base pointers 16B aligned.
// OOB rows/cols are index-CLAMPED during staging (duplicated data only ever
// contributes to outputs that the epilogue masks off).
// ---------------------------------------------------------------------------
template <int TB>
__global__ __launch_bounds__(256) void vit_gemm_bf16(
    const bf16* __restrict__ A,  long aO, long aI, int aC, int lda,
    const bf16* __restrict__ Bw, long bO, long bI, int bC, int ldb,
    float* __restrict__ outF, bf16* __restrict__ outH,
    long oO, long oI, int oC, int ldo,
    const float* __restrict__ bias,
    int M, int N, int K, int mode, float scale, int maskN)
{
    __shared__ bf16 As[2][BM][BK + KPAD];
    __shared__ bf16 Bs[2][BN][BK + KPAD];

    const int tid  = threadIdx.x;
    const int lane = tid & 31;
    const int wave = tid >> 5;
    const int wm   = (wave & 3) * 32;   // 4 waves along M
    const int wn   = (wave >> 2) * 64;  // 2 waves along N

    const int z = blockIdx.z;
    const long aOff = (long)(z / aC) * aO + (long)(z % aC) * aI;
    const long bOff = (long)(z / bC) * bO + (long)(z % bC) * bI;
    const long oOff = (long)(z / oC) * oO + (long)(z % oC) * oI;

    const int m0 = blockIdx.y * BM;
    const int n0 = blockIdx.x * BN;

    // Async-issue one BM x BK A tile into As[buf] (4 b128 copies per thread).
    auto issueA = [&](int buf, int kk) {
#pragma unroll
        for (int i = 0; i < 4; ++i) {
            int v  = tid + i * 256;       // 0..1023 b128 slots
            int r  = v >> 3;              // 0..127
            int kc = (v & 7) * 8;         // 0..56
            int gm = m0 + r; if (gm >= M) gm = M - 1;   // clamp (masked later)
            const bf16* gp = A + aOff + (long)gm * lda + (kk + kc);
            unsigned lp = (unsigned)(unsigned long long)(&As[buf][r][kc]);
            asm volatile("global_load_async_to_lds_b128 %0, %1, off"
                         :: "v"(lp), "v"(gp) : "memory");
        }
    };
    // Async-issue one BN x BK B tile (W[N,K] rows) into Bs[buf].
    auto issueB = [&](int buf, int kk) {
#pragma unroll
        for (int i = 0; i < 4; ++i) {
            int v  = tid + i * 256;
            int r  = v >> 3;
            int kc = (v & 7) * 8;
            int gn = n0 + r; if (gn >= N) gn = N - 1;
            const bf16* gp = Bw + bOff + (long)gn * ldb + (kk + kc);
            unsigned lp = (unsigned)(unsigned long long)(&Bs[buf][r][kc]);
            asm volatile("global_load_async_to_lds_b128 %0, %1, off"
                         :: "v"(lp), "v"(gp) : "memory");
        }
    };

    v8f acc[2][4];
#pragma unroll
    for (int i = 0; i < 2; ++i)
#pragma unroll
        for (int j = 0; j < 4; ++j)
#pragma unroll
            for (int e = 0; e < 8; ++e) acc[i][j][e] = 0.0f;

    // Fragment addressing per CDNA5 ISA layouts (wave32):
    //  A 16x32 bf16: lanes 0-15 hold row m=lane, K={0..7,16..23};
    //                lanes 16-31 hold same rows, K={8..15,24..31}.
    //  B 32x16 bf16: lanes 0-15 hold col n=lane, K={0..7,16..23}; lanes 16-31 K={8..15,24..31}.
    const int kbA = (lane < 16) ? 0 : 8;
    const int kbB = (lane < 16) ? 0 : 8;
    const int mr  = lane & 15;

    const int nk = K / BK;
    // prologue: stage tile 0
    issueA(0, 0);
    if constexpr (!TB) issueB(0, 0);

    for (int t = 0; t < nk; ++t) {
        const int cur = t & 1;
        const int k0  = t * BK;

        if (t + 1 < nk) {                       // issue next tile (overlaps compute)
            issueA(cur ^ 1, k0 + BK);
            if constexpr (!TB) issueB(cur ^ 1, k0 + BK);
        }
        if constexpr (TB) {
            // synchronous transpose stage: Bs[cur][n][k] = W[k0+k][n0+n]
            for (int i = 0; i < 32; ++i) {
                int e  = tid + i * 256;   // 0..8191
                int kk = e >> 7;          // 0..63
                int nn = e & 127;
                int gn = n0 + nn; if (gn >= N) gn = N - 1;
                Bs[cur][nn][kk] = Bw[bOff + (long)(k0 + kk) * ldb + gn];
            }
        }
        // drain current tile's async copies; leave next tile's in flight
        if (t + 1 < nk) {
            if constexpr (TB) asm volatile("s_wait_asynccnt 0x4" ::: "memory");
            else              asm volatile("s_wait_asynccnt 0x8" ::: "memory");
        } else {
            asm volatile("s_wait_asynccnt 0x0" ::: "memory");
        }
        __syncthreads();

        // ---- two WMMA K-steps (2 x 32) on the staged tile ----
#pragma unroll
        for (int s = 0; s < 2; ++s) {
            v16bf af[2], bfr[4];
#pragma unroll
            for (int i = 0; i < 2; ++i) {
                const bf16* row = &As[cur][wm + i * 16 + mr][s * 32];
                FragU u;
                u.q[0] = *(const uint4*)(row + kbA);
                u.q[1] = *(const uint4*)(row + kbA + 16);
                af[i] = u.v;
            }
#pragma unroll
            for (int j = 0; j < 4; ++j) {
                const bf16* row = &Bs[cur][wn + j * 16 + mr][s * 32];
                FragU u;
                u.q[0] = *(const uint4*)(row + kbB);
                u.q[1] = *(const uint4*)(row + kbB + 16);
                bfr[j] = u.v;
            }
#pragma unroll
            for (int i = 0; i < 2; ++i)
#pragma unroll
                for (int j = 0; j < 4; ++j)
                    acc[i][j] = __builtin_amdgcn_wmma_f32_16x16x32_bf16(
                        false, af[i], false, bfr[j], (short)0, acc[i][j], false, false);
        }
        __syncthreads();   // all waves done reading before buffer is re-filled
    }

    // ---- epilogue: C/D layout = lanes 0-15 n=lane m=e; lanes 16-31 n=lane-16 m=e+8
    const int mh = (lane >> 4) * 8;
    const int nl = lane & 15;
#pragma unroll
    for (int i = 0; i < 2; ++i) {
#pragma unroll
        for (int j = 0; j < 4; ++j) {
#pragma unroll
            for (int e = 0; e < 8; ++e) {
                int m = m0 + wm + i * 16 + mh + e;
                int n = n0 + wn + j * 16 + nl;
                if (m < M && n < N) {
                    float x = acc[i][j][e];
                    long idx = oOff + (long)m * ldo + n;
                    if (mode == 4) {
                        outF[idx] = (n < maskN) ? x * scale : -1.0e30f;
                    } else {
                        if (bias) x += bias[n];
                        if (mode == 0)      outF[idx] = x;
                        else if (mode == 1) outH[idx] = (bf16)x;
                        else if (mode == 2) {
                            float g = 0.5f * x * (1.0f + erff(x * 0.70710678118654752f));
                            outH[idx] = (bf16)g;
                        } else {            // mode 3: residual accumulate
                            outF[idx] = outF[idx] + x;
                        }
                    }
                }
            }
        }
    }
}

// ---------------------------------------------------------------------------
// Elementwise / reduction helpers
// ---------------------------------------------------------------------------
__global__ void vit_f32_to_bf16(const float* __restrict__ in, bf16* __restrict__ out, int n)
{
    for (int i = blockIdx.x * 256 + threadIdx.x; i < n; i += gridDim.x * 256)
        out[i] = (bf16)in[i];
}

// im2col: x[B,3,224,224] -> Xc[B*196, 768] (c*256 + py*16 + px ordering = conv_w layout)
__global__ void vit_im2col(const float* __restrict__ x, bf16* __restrict__ Xc, int total)
{
    for (int idx = blockIdx.x * 256 + threadIdx.x; idx < total; idx += gridDim.x * 256) {
        int row  = idx / 768;
        int cidx = idx - row * 768;
        int b  = row / 196;
        int t  = row - b * 196;
        int ty = t / 14, tx = t - ty * 14;
        int c  = cidx >> 8;
        int r  = cidx & 255;
        int py = r >> 4, px = r & 15;
        int iy = ty * 16 + py, ix = tx * 16 + px;
        Xc[idx] = (bf16)x[(((long)b * 3 + c) * 224 + iy) * 224 + ix];
    }
}

// h[b,t,d] = cls+pos (t=0) | tok + pos (1<=t<=196) | 0 (pad)
__global__ void vit_assemble(const float* __restrict__ tok, const float* __restrict__ cls,
                             const float* __restrict__ pos, float* __restrict__ h, int total)
{
    for (int idx = blockIdx.x * 256 + threadIdx.x; idx < total; idx += gridDim.x * 256) {
        int r = idx % (NP * DIM);
        int b = idx / (NP * DIM);
        int t = r / DIM;
        int d = r - t * DIM;
        float v;
        if (t == 0)        v = cls[d] + pos[d];
        else if (t <= 196) v = tok[((long)b * 196 + (t - 1)) * DIM + d] + pos[(long)t * DIM + d];
        else               v = 0.0f;
        h[idx] = v;
    }
}

// LayerNorm over D=768: one block (256 thr) per token row; writes bf16.
__global__ __launch_bounds__(256) void vit_layernorm(const float* __restrict__ x,
                                                     const float* __restrict__ w,
                                                     const float* __restrict__ b,
                                                     bf16* __restrict__ y)
{
    __shared__ float red[256];
    const long row = blockIdx.x;
    const float* xr = x + row * DIM;
    const int tid = threadIdx.x;
    float v0 = xr[tid], v1 = xr[tid + 256], v2 = xr[tid + 512];
    red[tid] = v0 + v1 + v2;
    __syncthreads();
    for (int off = 128; off > 0; off >>= 1) {
        if (tid < off) red[tid] += red[tid + off];
        __syncthreads();
    }
    float mean = red[0] * (1.0f / 768.0f);
    __syncthreads();
    float d0 = v0 - mean, d1 = v1 - mean, d2 = v2 - mean;
    red[tid] = d0 * d0 + d1 * d1 + d2 * d2;
    __syncthreads();
    for (int off = 128; off > 0; off >>= 1) {
        if (tid < off) red[tid] += red[tid + off];
        __syncthreads();
    }
    float rstd = rsqrtf(red[0] * (1.0f / 768.0f) + 1e-5f);
    bf16* yr = y + row * DIM;
    yr[tid]       = (bf16)(d0 * rstd * w[tid]       + b[tid]);
    yr[tid + 256] = (bf16)(d1 * rstd * w[tid + 256] + b[tid + 256]);
    yr[tid + 512] = (bf16)(d2 * rstd * w[tid + 512] + b[tid + 512]);
}

// Softmax over one score row of NP=256 (masked cols already -1e30) -> bf16 probs
__global__ __launch_bounds__(256) void vit_softmax(const float* __restrict__ S,
                                                   bf16* __restrict__ P)
{
    __shared__ float red[256];
    const long row = blockIdx.x;
    const float* sr = S + row * NP;
    const int tid = threadIdx.x;
    float v = sr[tid];
    red[tid] = v;
    __syncthreads();
    for (int off = 128; off > 0; off >>= 1) {
        if (tid < off) red[tid] = fmaxf(red[tid], red[tid + off]);
        __syncthreads();
    }
    float mx = red[0];
    __syncthreads();
    float e = __expf(v - mx);
    red[tid] = e;
    __syncthreads();
    for (int off = 128; off > 0; off >>= 1) {
        if (tid < off) red[tid] += red[tid + off];
        __syncthreads();
    }
    P[row * NP + tid] = (bf16)(e / red[0]);
}

// ---------------------------------------------------------------------------
// Host launcher
// ---------------------------------------------------------------------------
extern "C" void kernel_launch(void* const* d_in, const int* in_sizes, int n_in,
                              void* d_out, int out_size, void* d_ws, size_t ws_size,
                              hipStream_t stream)
{
    const float* x      = (const float*)d_in[0];
    const float* conv_w = (const float*)d_in[1];
    const float* conv_b = (const float*)d_in[2];
    const float* cls    = (const float*)d_in[3];
    const float* pos    = (const float*)d_in[4];
    const float* ln1w   = (const float*)d_in[5];
    const float* ln1b   = (const float*)d_in[6];
    const float* qkvw   = (const float*)d_in[7];
    const float* qkvb   = (const float*)d_in[8];
    const float* projw  = (const float*)d_in[9];
    const float* projb  = (const float*)d_in[10];
    const float* ln2w   = (const float*)d_in[11];
    const float* ln2b   = (const float*)d_in[12];
    const float* fc1w   = (const float*)d_in[13];
    const float* fc1b   = (const float*)d_in[14];
    const float* fc2w   = (const float*)d_in[15];
    const float* fc2b   = (const float*)d_in[16];
    const float* lnfw   = (const float*)d_in[17];
    const float* lnfb   = (const float*)d_in[18];
    const float* headw  = (const float*)d_in[19];
    const float* headb  = (const float*)d_in[20];
    float* out = (float*)d_out;

    // ---- workspace layout (256B aligned) ----
    char* ws = (char*)d_ws;
    size_t off = 0;
    auto alloc = [&](size_t bytes) { size_t o = off; off = (off + bytes + 255) & ~(size_t)255; return o; };
    const size_t o_wqkv  = alloc((size_t)NLAYER * 3 * DIM * DIM * 2);
    const size_t o_wproj = alloc((size_t)NLAYER * DIM * DIM * 2);
    const size_t o_wfc1  = alloc((size_t)NLAYER * MLPD * DIM * 2);
    const size_t o_wfc2  = alloc((size_t)NLAYER * DIM * MLPD * 2);
    const size_t o_wconv = alloc((size_t)DIM * 768 * 2);
    const size_t o_whead = alloc((size_t)NCLS * DIM * 2);
    const size_t o_h     = alloc((size_t)MT * DIM * 4);
    const size_t o_y     = alloc((size_t)MT * DIM * 2);
    const size_t o_qkv   = alloc((size_t)MT * 3 * DIM * 2);
    const size_t o_S     = alloc((size_t)BSZ * NH * NP * NP * 4);
    const size_t o_P     = alloc((size_t)BSZ * NH * NP * NP * 2);
    const size_t o_ob    = alloc((size_t)MT * DIM * 2);
    const size_t o_m     = alloc((size_t)MT * MLPD * 2);
    // patch-embed scratch aliased into the (later-used) score buffer
    const size_t o_Xc  = o_S;
    const size_t o_tok = o_S + ((((size_t)BSZ * 196 * 768 * 2) + 255) & ~(size_t)255);

    bf16*  wqkv  = (bf16*)(ws + o_wqkv);
    bf16*  wproj = (bf16*)(ws + o_wproj);
    bf16*  wfc1  = (bf16*)(ws + o_wfc1);
    bf16*  wfc2  = (bf16*)(ws + o_wfc2);
    bf16*  wconv = (bf16*)(ws + o_wconv);
    bf16*  whead = (bf16*)(ws + o_whead);
    float* h     = (float*)(ws + o_h);
    bf16*  y     = (bf16*)(ws + o_y);
    bf16*  qkv   = (bf16*)(ws + o_qkv);
    float* Sf    = (float*)(ws + o_S);
    bf16*  Pb    = (bf16*)(ws + o_P);
    bf16*  ob    = (bf16*)(ws + o_ob);
    bf16*  mb    = (bf16*)(ws + o_m);
    bf16*  Xc    = (bf16*)(ws + o_Xc);
    float* tokf  = (float*)(ws + o_tok);

    auto cvt = [&](const float* src, bf16* dst, long n) {
        int nn = (int)n;
        vit_f32_to_bf16<<<(nn + 255) / 256, 256, 0, stream>>>(src, dst, nn);
    };
    auto gemm = [&](const bf16* A, long aO, long aI, int aC, int lda,
                    const bf16* Bw, long bO, long bI, int bC, int ldb, int tB,
                    float* oF, bf16* oH, long oO, long oI, int oC, int ldo,
                    const float* bias, int M, int N, int K, int mode,
                    float scl, int maskN, int batch) {
        dim3 g((N + BN - 1) / BN, (M + BM - 1) / BM, batch);
        if (tB)
            vit_gemm_bf16<1><<<g, dim3(256), 0, stream>>>(A, aO, aI, aC, lda,
                                                          Bw, bO, bI, bC, ldb,
                                                          oF, oH, oO, oI, oC, ldo,
                                                          bias, M, N, K, mode, scl, maskN);
        else
            vit_gemm_bf16<0><<<g, dim3(256), 0, stream>>>(A, aO, aI, aC, lda,
                                                          Bw, bO, bI, bC, ldb,
                                                          oF, oH, oO, oI, oC, ldo,
                                                          bias, M, N, K, mode, scl, maskN);
    };

    // ---- weight conversion fp32 -> bf16 ----
    cvt(qkvw,  wqkv,  (long)NLAYER * 3 * DIM * DIM);
    cvt(projw, wproj, (long)NLAYER * DIM * DIM);
    cvt(fc1w,  wfc1,  (long)NLAYER * MLPD * DIM);
    cvt(fc2w,  wfc2,  (long)NLAYER * DIM * MLPD);
    cvt(conv_w, wconv, (long)DIM * 768);
    cvt(headw, whead, (long)NCLS * DIM);

    // ---- patch embedding: im2col + GEMM + assemble ----
    {
        int total = BSZ * 196 * 768;
        vit_im2col<<<(total + 255) / 256, 256, 0, stream>>>(x, Xc, total);
        gemm(Xc, 0, 0, 1, 768, wconv, 0, 0, 1, 768, 0,
             tokf, nullptr, 0, 0, 1, DIM, conv_b,
             BSZ * 196, DIM, 768, /*mode*/0, 1.0f, 0, 1);
        int tot2 = BSZ * NP * DIM;
        vit_assemble<<<(tot2 + 255) / 256, 256, 0, stream>>>(tokf, cls, pos, h, tot2);
    }

    // ---- transformer blocks ----
    const float attnScale = 0.125f;  // HD^-0.5 = 1/8
    for (int l = 0; l < NLAYER; ++l) {
        vit_layernorm<<<(int)MT, 256, 0, stream>>>(h, ln1w + l * DIM, ln1b + l * DIM, y);

        // qkv = y @ qkv_w^T + qkv_b   -> bf16 [MT, 2304]
        gemm(y, 0, 0, 1, DIM,
             wqkv + (long)l * 3 * DIM * DIM, 0, 0, 1, DIM, 0,
             nullptr, qkv, 0, 0, 1, 3 * DIM,
             qkvb + (long)l * 3 * DIM, (int)MT, 3 * DIM, DIM, /*mode*/1, 1.0f, 0, 1);

        // scores[b,h] = Q @ K^T * scale, cols >= 197 masked   (batch = B*H)
        gemm(qkv,            (long)NP * 3 * DIM, HDIM, NH, 3 * DIM,
             qkv + DIM,      (long)NP * 3 * DIM, HDIM, NH, 3 * DIM, 0,
             Sf, nullptr,    (long)NP * NP, 0, 1, NP,
             nullptr, NP, NP, HDIM, /*mode*/4, attnScale, NT, BSZ * NH);

        vit_softmax<<<BSZ * NH * NP, 256, 0, stream>>>(Sf, Pb);

        // o[b,h] = P @ V  -> bf16 laid out as [b, t, h*64+d] in [MT, 768]
        gemm(Pb,             (long)NP * NP, 0, 1, NP,
             qkv + 2 * DIM,  (long)NP * 3 * DIM, HDIM, NH, 3 * DIM, /*transB*/1,
             nullptr, ob,    (long)NP * DIM, HDIM, NH, DIM,
             nullptr, NP, HDIM, NP, /*mode*/1, 1.0f, 0, BSZ * NH);

        // h += o @ proj_w^T + proj_b
        gemm(ob, 0, 0, 1, DIM,
             wproj + (long)l * DIM * DIM, 0, 0, 1, DIM, 0,
             h, nullptr, 0, 0, 1, DIM,
             projb + (long)l * DIM, (int)MT, DIM, DIM, /*mode*/3, 1.0f, 0, 1);

        vit_layernorm<<<(int)MT, 256, 0, stream>>>(h, ln2w + l * DIM, ln2b + l * DIM, y);

        // m = gelu(y @ fc1_w^T + fc1_b)  -> bf16 [MT, 3072]
        gemm(y, 0, 0, 1, DIM,
             wfc1 + (long)l * MLPD * DIM, 0, 0, 1, DIM, 0,
             nullptr, mb, 0, 0, 1, MLPD,
             fc1b + (long)l * MLPD, (int)MT, MLPD, DIM, /*mode*/2, 1.0f, 0, 1);

        // h += m @ fc2_w^T + fc2_b
        gemm(mb, 0, 0, 1, MLPD,
             wfc2 + (long)l * DIM * MLPD, 0, 0, 1, MLPD, 0,
             h, nullptr, 0, 0, 1, DIM,
             fc2b + (long)l * DIM, (int)MT, DIM, MLPD, /*mode*/3, 1.0f, 0, 1);
    }

    // ---- final LN + classification head (cls rows only: lda = NP*DIM) ----
    vit_layernorm<<<(int)MT, 256, 0, stream>>>(h, lnfw, lnfb, y);
    gemm(y, 0, 0, 1, NP * DIM,
         whead, 0, 0, 1, DIM, 0,
         out, nullptr, 0, 0, 1, NCLS,
         headb, BSZ, NCLS, DIM, /*mode*/0, 1.0f, 0, 1);
}
